// LocalRNN_69741678952750
// MI455X (gfx1250) — matrix operand order repaired and checked by reading, not compiled, not run on hardware
//
#include <hip/hip_runtime.h>

#define BB 16
#define LL 2048
#define DD 128
#define KS 8
#define GATES 384          // 3*D
#define BL (BB*LL)         // 32768 windows
#define WPW 32             // windows per wave in the GRU kernel (2 x 16-tile)

typedef _Float16 h16v __attribute__((ext_vector_type(16)));
typedef _Float16 h8v  __attribute__((ext_vector_type(8)));
typedef _Float16 h4v  __attribute__((ext_vector_type(4)));
typedef float    f8v  __attribute__((ext_vector_type(8)));
typedef float    f4v  __attribute__((ext_vector_type(4)));

// Dynamic LDS layout:
//   [0, 96KB)  : W A-fragments, pre-swizzled: [mtile(24)][kblk(4)][lane(32)] x 32B (16 halves)
//   [96KB, ..) : per-wave h/x tiles: 16 rows x 128 cols f16 (B-operand friendly, row-major)
#define LDS_W_HALVES    (24*4*32*16)     // 49152 halves = 96 KB
#define LDS_TILE_HALVES (16*128)         // 2048 halves = 4 KB per 16-window tile
#define LDS_GX_BYTES   (LDS_W_HALVES*2 + 8*LDS_TILE_HALVES*2)    // 128 KB
#define LDS_GRU_BYTES  (LDS_W_HALVES*2 + 16*LDS_TILE_HALVES*2)   // 160 KB

// fast activations: v_exp_f32 / v_rcp_f32 TRANS ops (co-execute with XDL WMMA pipe)
__device__ __forceinline__ float sigmoid_fast(float x) {
  return __builtin_amdgcn_rcpf(1.0f + __expf(-x));
}
__device__ __forceinline__ float tanh_fast(float x) {
  const float e = __expf(-2.0f * x);
  return (1.0f - e) * __builtin_amdgcn_rcpf(1.0f + e);
}

// A-operand (16x32 f16) half-index -> K mapping within a 32-wide K block.
// lane s = lane>>4:  halves 0..15 -> K = i + 16*s (contiguous, matches B fragments)
__device__ __forceinline__ int k32_of(int i, int s) {
  return i + 16*s;
}

// Build the A-operand swizzled f16 copy of a [384 x 128] f32 weight matrix in LDS.
__device__ __forceinline__ void build_w_frags(const float* __restrict__ W, _Float16* lds) {
  const int tid = threadIdx.x;
  for (int f = tid; f < 24*4*32; f += 256) {
    const int mt = f >> 7;           // / (4*32)
    const int kb = (f >> 5) & 3;
    const int ln = f & 31;
    const int m  = (mt << 4) + (ln & 15);
    const int s  = ln >> 4;
    _Float16* dst = lds + (size_t)f * 16;
    const float* src = W + (size_t)m * DD + kb * 32;
    #pragma unroll
    for (int i = 0; i < 16; ++i) dst[i] = (_Float16)src[k32_of(i, s)];
  }
}

// Writes the "padding" row of Gx: pure bias (x == 0): b_ih (+ b_hh for r,z parts).
__global__ void __launch_bounds__(384) bias_row_kernel(const float* __restrict__ b_ih,
                                                       const float* __restrict__ b_hh,
                                                       float* __restrict__ Gx) {
  const int g = threadIdx.x;
  if (g < GATES) Gx[(size_t)BL * GATES + g] = b_ih[g] + (g < 256 ? b_hh[g] : 0.0f);
}

// Gx[row][gate] = x[row,:] @ W_ih[gate,:] + b_ih[gate] (+ b_hh[gate] for r,z gates)
__global__ void __launch_bounds__(256) gx_kernel(const float* __restrict__ x,
                                                 const float* __restrict__ W_ih,
                                                 const float* __restrict__ b_ih,
                                                 const float* __restrict__ b_hh,
                                                 float* __restrict__ Gx) {
  extern __shared__ __attribute__((aligned(32))) _Float16 lds[];
  build_w_frags(W_ih, lds);

  const int tid  = threadIdx.x;
  const int wave = tid >> 5;
  const int lane = tid & 31;
  const int w    = lane & 15;
  const int hi   = lane >> 4;
  const int row0 = (blockIdx.x * 8 + wave) * 16;

  // stage this wave's 16x128 x tile as f16 row-major in LDS
  _Float16* xt = lds + LDS_W_HALVES + (size_t)wave * LDS_TILE_HALVES;
  for (int q = lane; q < 16*32; q += 32) {
    const int r  = q >> 5;
    const int c4 = (q & 31) * 4;
    f4v v = *(const f4v*)(x + (size_t)(row0 + r) * DD + c4);
    h4v h; h.x = (_Float16)v.x; h.y = (_Float16)v.y; h.z = (_Float16)v.z; h.w = (_Float16)v.w;
    *(h4v*)(xt + r * DD + c4) = h;
  }
  __syncthreads();

  // B-fragments of x (K = 128 -> 4 fragments), kept in registers
  h16v bf[4];
  #pragma unroll
  for (int kb = 0; kb < 4; ++kb)
    bf[kb] = *(const h16v*)(xt + w * DD + kb * 32 + hi * 16);

  const size_t xrow = (size_t)(row0 + w);
  #pragma unroll 4
  for (int g = 0; g < 24; ++g) {
    f8v acc = {};
    #pragma unroll
    for (int kb = 0; kb < 4; ++kb) {
      h16v af = *(const h16v*)(lds + ((size_t)(g * 4 + kb) * 32 + lane) * 16);
      acc = __builtin_amdgcn_wmma_f32_16x16x32_f16(false, af, false, bf[kb],
                                                   (short)0, acc, false, false);
    }
    const int c0 = g * 16 + hi * 8;          // 8 consecutive gate columns per lane
    f8v bias;
    {
      f4v b0 = *(const f4v*)(b_ih + c0);
      f4v b1 = *(const f4v*)(b_ih + c0 + 4);
      if (c0 < 256) {                        // fold b_hh into r,z gate biases
        b0 += *(const f4v*)(b_hh + c0);
        b1 += *(const f4v*)(b_hh + c0 + 4);
      }
      bias[0]=b0.x; bias[1]=b0.y; bias[2]=b0.z; bias[3]=b0.w;
      bias[4]=b1.x; bias[5]=b1.y; bias[6]=b1.z; bias[7]=b1.w;
    }
    f8v o = acc + bias;
    *(f8v*)(Gx + xrow * GATES + c0) = o;
  }
}

// elementwise GRU update for one 16-window tile / one 16-column group
__device__ __forceinline__ void gru_elem(const float* __restrict__ grow,
                                         const float* __restrict__ b_hh,
                                         _Float16* __restrict__ ht,
                                         float* __restrict__ out,
                                         size_t outrow, int c0, int w, bool last,
                                         const f8v& ar, const f8v& az, const f8v& an) {
  f8v gr = *(const f8v*)(grow + c0);          // b_ih + b_hh_r already folded in
  f8v gz = *(const f8v*)(grow + 128 + c0);
  f8v gn = *(const f8v*)(grow + 256 + c0);    // b_ih only
  f8v bn = *(const f8v*)(b_hh + 256 + c0);    // b_hh_n applied inside r*(.)
  h8v hold = *(const h8v*)(ht + w * DD + c0);

  f8v hnew;
  #pragma unroll
  for (int v = 0; v < 8; ++v) {
    const float r_ = sigmoid_fast(gr[v] + ar[v]);
    const float z_ = sigmoid_fast(gz[v] + az[v]);
    const float n_ = tanh_fast(gn[v] + r_ * (an[v] + bn[v]));
    hnew[v] = (1.0f - z_) * n_ + z_ * (float)hold[v];
  }
  if (!last) {
    h8v hh;
    #pragma unroll
    for (int v = 0; v < 8; ++v) hh[v] = (_Float16)hnew[v];
    *(h8v*)(ht + w * DD + c0) = hh;
  } else {
    *(f8v*)(out + outrow * DD + c0) = hnew;
  }
}

// 8-step GRU recurrence; each wave owns 32 windows (two 16-window B-tiles),
// so every W A-fragment LDS load feeds two WMMAs.
__global__ void __launch_bounds__(256) gru_kernel(const float* __restrict__ Gx,
                                                  const float* __restrict__ W_hh,
                                                  const float* __restrict__ b_hh,
                                                  float* __restrict__ out) {
  extern __shared__ __attribute__((aligned(32))) _Float16 lds[];
  build_w_frags(W_hh, lds);

  const int tid  = threadIdx.x;
  const int wave = tid >> 5;
  const int lane = tid & 31;
  const int w    = lane & 15;
  const int hi   = lane >> 4;
  const int row0 = (blockIdx.x * 8 + wave) * WPW;
  const int b    = row0 >> 11;               // / LL  (32-window tiles never straddle batches)
  const int l0   = row0 & (LL - 1);
  const int la   = l0 + w;                   // lane's window in tile 0
  const int lb   = l0 + 16 + w;              // lane's window in tile 1

  _Float16* ht0 = lds + LDS_W_HALVES + (size_t)(wave * 2) * LDS_TILE_HALVES;
  _Float16* ht1 = ht0 + LDS_TILE_HALVES;
  // h = 0 for both tiles (contiguous)
  for (int q = lane; q < 2 * LDS_TILE_HALVES / 8; q += 32) {
    h8v z = {};
    ((h8v*)ht0)[q] = z;
  }
  __syncthreads();

  for (int t = 0; t < KS; ++t) {
    // B-fragments of h for both tiles (K = 128), in registers for all WMMAs this step
    h16v hb0[4], hb1[4];
    #pragma unroll
    for (int kb = 0; kb < 4; ++kb) {
      hb0[kb] = *(const h16v*)(ht0 + w * DD + kb * 32 + hi * 16);
      hb1[kb] = *(const h16v*)(ht1 + w * DD + kb * 32 + hi * 16);
    }

    const int lpa = la + t - (KS - 1);
    const int lpb = lb + t - (KS - 1);
    const float* ga = (lpa >= 0) ? (Gx + ((size_t)b * LL + lpa) * GATES)
                                 : (Gx + (size_t)BL * GATES);   // pure-bias row
    const float* gb = (lpb >= 0) ? (Gx + ((size_t)b * LL + lpb) * GATES)
                                 : (Gx + (size_t)BL * GATES);
    if (t + 1 < KS) {
      if (lpa + 1 >= 0) __builtin_prefetch(Gx + ((size_t)b * LL + (lpa + 1)) * GATES, 0, 1);
      __builtin_prefetch(Gx + ((size_t)b * LL + (lpb + 1)) * GATES, 0, 1);
    }

    const bool last = (t == KS - 1);
    for (int j = 0; j < 8; ++j) {            // 8 groups of 16 hidden columns
      f8v ar0 = {}, az0 = {}, an0 = {};
      f8v ar1 = {}, az1 = {}, an1 = {};
      #pragma unroll
      for (int kb = 0; kb < 4; ++kb) {
        h16v aR = *(const h16v*)(lds + ((size_t)((j      ) * 4 + kb) * 32 + lane) * 16);
        h16v aZ = *(const h16v*)(lds + ((size_t)((j +  8 ) * 4 + kb) * 32 + lane) * 16);
        h16v aN = *(const h16v*)(lds + ((size_t)((j + 16 ) * 4 + kb) * 32 + lane) * 16);
        ar0 = __builtin_amdgcn_wmma_f32_16x16x32_f16(false, aR, false, hb0[kb], (short)0, ar0, false, false);
        ar1 = __builtin_amdgcn_wmma_f32_16x16x32_f16(false, aR, false, hb1[kb], (short)0, ar1, false, false);
        az0 = __builtin_amdgcn_wmma_f32_16x16x32_f16(false, aZ, false, hb0[kb], (short)0, az0, false, false);
        az1 = __builtin_amdgcn_wmma_f32_16x16x32_f16(false, aZ, false, hb1[kb], (short)0, az1, false, false);
        an0 = __builtin_amdgcn_wmma_f32_16x16x32_f16(false, aN, false, hb0[kb], (short)0, an0, false, false);
        an1 = __builtin_amdgcn_wmma_f32_16x16x32_f16(false, aN, false, hb1[kb], (short)0, an1, false, false);
      }
      const int c0 = j * 16 + hi * 8;        // 8 consecutive h-columns per lane
      gru_elem(ga, b_hh, ht0, out, (size_t)b * LL + la, c0, w, last, ar0, az0, an0);
      gru_elem(gb, b_hh, ht1, out, (size_t)b * LL + lb, c0, w, last, ar1, az1, an1);
    }
    __syncthreads();   // order h LDS write-back vs next step's cross-lane B-frag loads
  }
}

extern "C" void kernel_launch(void* const* d_in, const int* in_sizes, int n_in,
                              void* d_out, int out_size, void* d_ws, size_t ws_size,
                              hipStream_t stream) {
  const float* x    = (const float*)d_in[0];
  const float* W_ih = (const float*)d_in[1];
  const float* W_hh = (const float*)d_in[2];
  const float* b_ih = (const float*)d_in[3];
  const float* b_hh = (const float*)d_in[4];
  float* out = (float*)d_out;
  float* Gx  = (float*)d_ws;                 // (BL+1)*384 f32 ~= 48.2 MB scratch

  bias_row_kernel<<<1, 384, 0, stream>>>(b_ih, b_hh, Gx);
  gx_kernel<<<BL / 128, 256, LDS_GX_BYTES, stream>>>(x, W_ih, b_ih, b_hh, Gx);
  gru_kernel<<<BL / 256, 256, LDS_GRU_BYTES, stream>>>(Gx, W_hh, b_hh, out);
}
